// Multi_Heads_Self_Attn_Q_KV_dot_prod_83021717832741
// MI455X (gfx1250) — matrix operand order repaired and checked by reading, not compiled, not run on hardware
//
#include <hip/hip_runtime.h>
#include <hip/hip_bf16.h>

// ---------------------------------------------------------------------------
// Multi-head *linear* attention, algebraically collapsed:
//   Y[b] = (Wo · BD(M_h^T) · Wq) · Xq[b],  M_h = Wk_h · (Xkv Xkv^T) · Wv_h^T / 1728
// Big matmuls (Gram + output GEMM) use v_wmma_f32_16x16x32_bf16 (wave32).
// ---------------------------------------------------------------------------

typedef __attribute__((ext_vector_type(16))) __bf16       bf16x16;
typedef __attribute__((ext_vector_type(8)))  float        f32x8;
typedef __attribute__((ext_vector_type(4)))  unsigned int u32x4;

union FragU { bf16x16 v; u32x4 q[2]; };

static constexpr int B_ = 4;     // batch
static constexpr int CH = 256;   // channels (= INNER = N_Q = N_KV)
static constexpr int NS = 1728;  // spatial 12*12*12
static constexpr int HD = 8;     // heads
static constexpr int DK = 32;    // head dim

// workspace byte offsets (all 256B aligned)
static constexpr size_t OFF_KVB  = 0;                                    // bf16 [B,CH,NS]
static constexpr size_t OFF_XQT  = OFF_KVB  + (size_t)B_*CH*NS*2;        // bf16 [B,NS,CH]
static constexpr size_t OFF_G    = OFF_XQT  + (size_t)B_*CH*NS*2;        // f32  [B,CH,CH]
static constexpr size_t OFF_T1   = OFF_G    + (size_t)B_*CH*CH*4;        // f32  [B,CH,CH]
static constexpr size_t OFF_M    = OFF_T1   + (size_t)B_*CH*CH*4;        // f32  [B,HD,DK,DK]
static constexpr size_t OFF_P    = OFF_M    + (size_t)B_*HD*DK*DK*4;     // f32  [B,CH,CH]
static constexpr size_t OFF_WEFF = OFF_P    + (size_t)B_*CH*CH*4;        // bf16 [B,CH,CH]

// ---------------- prep: f32 -> bf16 row-major copy of Xkv --------------------
__global__ void k_convert_kv(const float* __restrict__ xkv, __bf16* __restrict__ kvb) {
    int i = blockIdx.x * blockDim.x + threadIdx.x;      // grid sized exactly B*CH*NS
    kvb[i] = (__bf16)xkv[i];
}

// ---------------- prep: f32 [B,CH,NS] -> bf16 transposed [B,NS,CH] ----------
__global__ void k_transpose_q(const float* __restrict__ xq, __bf16* __restrict__ xqT) {
    int i = blockIdx.x * blockDim.x + threadIdx.x;      // over B*NS*CH (output idx)
    int c = i & (CH - 1);
    int n = (i >> 8) % NS;
    int b = i / (CH * NS);
    xqT[i] = (__bf16)xq[((size_t)b * CH + c) * NS + n];
}

// -------- fragment loaders (CDNA5 16x16x32 bf16 layouts, ISA 7.12.2) --------
__device__ __forceinline__ bf16x16 load_frag_A(const __bf16* row, int kb, int half) {
    // A[m, k]: elems 0..7 -> k = kb + half*8 + {0..7};  8..15 -> k + 16
    const __bf16* p = row + kb + half * 8;
    FragU f;
    f.q[0] = *reinterpret_cast<const u32x4*>(p);
    f.q[1] = *reinterpret_cast<const u32x4*>(p + 16);
    return f.v;
}
__device__ __forceinline__ bf16x16 load_frag_B(const __bf16* row, int kb, int half) {
    // B[k, n] with source row = n-major: elems 0..7 -> k = kb + half*8 + {0..7};  8..15 -> k + 16
    const __bf16* p = row + kb + half * 8;
    FragU f;
    f.q[0] = *reinterpret_cast<const u32x4*>(p);
    f.q[1] = *reinterpret_cast<const u32x4*>(p + 16);
    return f.v;
}

// ---------------- G[b] = Xkv[b] · Xkv[b]^T  (bf16 WMMA, f32 acc) ------------
// 1 wave -> 16x64 tile of G. 256 waves total (4 b * 16 mt * 4 nb).
__global__ void k_gram_wmma(const __bf16* __restrict__ kvb, float* __restrict__ G) {
    int w    = (blockIdx.x * blockDim.x + threadIdx.x) >> 5;
    int lane = threadIdx.x & 31;
    int b    = w >> 6;
    int mt   = (w >> 2) & 15;
    int nb   = w & 3;
    int half = lane >> 4;
    int l15  = lane & 15;
    int mBase = mt * 16, nBase = nb * 64;

    const __bf16* base = kvb + (size_t)b * CH * NS;
    const __bf16* arow = base + (size_t)(mBase + l15) * NS;

    f32x8 acc[4] = {};
    for (int kb = 0; kb < NS; kb += 32) {
        bf16x16 a = load_frag_A(arow, kb, half);
        #pragma unroll
        for (int j = 0; j < 4; ++j) {
            const __bf16* brow = base + (size_t)(nBase + j * 16 + l15) * NS;
            bf16x16 bfr = load_frag_B(brow, kb, half);
            acc[j] = __builtin_amdgcn_wmma_f32_16x16x32_bf16(
                false, a, false, bfr, (short)0, acc[j], false, false);
        }
    }
    float* g = G + (size_t)b * CH * CH;
    int m0 = mBase + half * 8;
    #pragma unroll
    for (int j = 0; j < 4; ++j) {
        int n = nBase + j * 16 + l15;
        #pragma unroll
        for (int r = 0; r < 8; ++r)
            g[(size_t)(m0 + r) * CH + n] = acc[j][r];
    }
}

// ---------------- tiny exact-f32 chain for W_eff ----------------------------
__global__ void k_t1(const float* __restrict__ Wk, const float* __restrict__ G,
                     float* __restrict__ T1) {                 // T1 = Wk · G
    int idx = blockIdx.x * blockDim.x + threadIdx.x;           // B*CH*CH
    int c = idx & 255, i = (idx >> 8) & 255, b = idx >> 16;
    const float* g = G + (size_t)b * CH * CH;
    float s = 0.f;
    for (int k = 0; k < CH; ++k) s += Wk[i * CH + k] * g[(size_t)k * CH + c];
    T1[idx] = s;
}
__global__ void k_m(const float* __restrict__ T1, const float* __restrict__ Wv,
                    float* __restrict__ M) {                   // M_h = T1_h · Wv_h^T / NS
    int idx = blockIdx.x * blockDim.x + threadIdx.x;           // B*HD*DK*DK
    int d2 = idx & 31, d1 = (idx >> 5) & 31, h = (idx >> 10) & 7, b = idx >> 13;
    const float* t = T1 + ((size_t)b * CH + h * DK + d1) * CH;
    const float* wv = Wv + (size_t)(h * DK + d2) * CH;
    float s = 0.f;
    for (int c = 0; c < CH; ++c) s += t[c] * wv[c];
    M[idx] = s * (1.0f / (float)NS);
}
__global__ void k_p(const float* __restrict__ M, const float* __restrict__ Wq,
                    float* __restrict__ P) {                   // P = BD(M^T) · Wq
    int idx = blockIdx.x * blockDim.x + threadIdx.x;           // B*CH*CH
    int c = idx & 255, row = (idx >> 8) & 255, b = idx >> 16;
    int h = row >> 5, i = row & 31;
    const float* m = M + ((size_t)b * HD + h) * DK * DK;
    float s = 0.f;
    for (int j = 0; j < DK; ++j) s += m[j * DK + i] * Wq[(h * DK + j) * CH + c];
    P[idx] = s;
}
__global__ void k_weff(const float* __restrict__ Wo, const float* __restrict__ P,
                       __bf16* __restrict__ weff) {            // W_eff = Wo · P (-> bf16)
    int idx = blockIdx.x * blockDim.x + threadIdx.x;           // B*CH*CH
    int c = idx & 255, o = (idx >> 8) & 255, b = idx >> 16;
    const float* p = P + (size_t)b * CH * CH;
    float s = 0.f;
    for (int k = 0; k < CH; ++k) s += Wo[o * CH + k] * p[(size_t)k * CH + c];
    weff[idx] = (__bf16)s;
}

// -------- Y[b] = W_eff[b] · Xq[b]; out = relu(Y + xq)  (bf16 WMMA) ----------
// 1 wave -> 16x64 tile. 1728 waves total (4 b * 16 mt * 27 nb).
__global__ void k_out_wmma(const __bf16* __restrict__ weff, const __bf16* __restrict__ xqT,
                           const float* __restrict__ xq, float* __restrict__ out) {
    int w    = (blockIdx.x * blockDim.x + threadIdx.x) >> 5;
    int lane = threadIdx.x & 31;
    int b    = w / 432;
    int rem  = w % 432;
    int mt   = rem / 27;
    int nb   = rem % 27;
    int half = lane >> 4;
    int l15  = lane & 15;
    int mBase = mt * 16, nBase = nb * 64;

    const __bf16* abase = weff + (size_t)b * CH * CH;
    const __bf16* bbase = xqT + (size_t)b * NS * CH;
    const __bf16* arow  = abase + (size_t)(mBase + l15) * CH;

    f32x8 acc[4] = {};
    #pragma unroll
    for (int kb = 0; kb < CH; kb += 32) {
        bf16x16 a = load_frag_A(arow, kb, half);
        #pragma unroll
        for (int j = 0; j < 4; ++j) {
            const __bf16* brow = bbase + (size_t)(nBase + j * 16 + l15) * CH;
            bf16x16 bfr = load_frag_B(brow, kb, half);
            acc[j] = __builtin_amdgcn_wmma_f32_16x16x32_bf16(
                false, a, false, bfr, (short)0, acc[j], false, false);
        }
    }
    const float* xr = xq + (size_t)b * CH * NS;
    float*       yr = out + (size_t)b * CH * NS;
    int m0 = mBase + half * 8;
    #pragma unroll
    for (int j = 0; j < 4; ++j) {
        int n = nBase + j * 16 + l15;
        #pragma unroll
        for (int r = 0; r < 8; ++r) {
            size_t idx = (size_t)(m0 + r) * NS + n;
            float y = acc[j][r] + xr[idx];
            yr[idx] = y > 0.f ? y : 0.f;
        }
    }
}

// ---------------------------------------------------------------------------
extern "C" void kernel_launch(void* const* d_in, const int* in_sizes, int n_in,
                              void* d_out, int out_size, void* d_ws, size_t ws_size,
                              hipStream_t stream) {
    const float* xq  = (const float*)d_in[0];
    const float* xkv = (const float*)d_in[1];
    const float* Wq  = (const float*)d_in[2];
    const float* Wk  = (const float*)d_in[3];
    const float* Wv  = (const float*)d_in[4];
    const float* Wo  = (const float*)d_in[5];
    float* out = (float*)d_out;

    char* ws = (char*)d_ws;
    __bf16* kvb  = (__bf16*)(ws + OFF_KVB);
    __bf16* xqT  = (__bf16*)(ws + OFF_XQT);
    float*  G    = (float*)(ws + OFF_G);
    float*  T1   = (float*)(ws + OFF_T1);
    float*  M    = (float*)(ws + OFF_M);
    float*  P    = (float*)(ws + OFF_P);
    __bf16* weff = (__bf16*)(ws + OFF_WEFF);

    // prep (B*CH*NS = 1,769,472 elements)
    k_convert_kv <<<6912, 256, 0, stream>>>(xkv, kvb);
    k_transpose_q<<<6912, 256, 0, stream>>>(xq, xqT);

    // Gram matrix: 256 waves = 64 blocks * 4 waves
    k_gram_wmma<<<64, 128, 0, stream>>>(kvb, G);

    // tiny f32 chain (B*CH*CH = 262,144 threads; B*HD*DK*DK = 32,768 threads)
    k_t1  <<<1024, 256, 0, stream>>>(Wk, G, T1);
    k_m   <<<128, 256, 0, stream>>>(T1, Wv, M);
    k_p   <<<1024, 256, 0, stream>>>(M, Wq, P);
    k_weff<<<1024, 256, 0, stream>>>(Wo, P, weff);

    // fused output GEMM + residual + ReLU: 1728 waves = 432 blocks * 4 waves
    k_out_wmma<<<432, 128, 0, stream>>>(weff, xqT, xq, out);
}
